// QuantizedLinear_57561151701240
// MI455X (gfx1250) — matrix operand order repaired and checked by reading, not compiled, not run on hardware
//
#include <hip/hip_runtime.h>

// ---------------------------------------------------------------------------
// QuantizedLinear for MI455X (gfx1250):  out = x @ (qw * scale[:,None])^T + bias
//   x:      [4, 4096]    fp32
//   qw:     [11008,4096] int32 (values fit int8 -> exact in f16)
//   scales: [11008]      fp32
//   bias:   [11008]      fp32
//   out:    [4, 11008]   fp32
//
// Roofline: 180 MB weight stream @ 23.3 TB/s => ~7.7 us floor; 0.36 GFLOP is
// negligible -> pure bandwidth problem. WMMA f32_16x16x32_f16 does the MACs
// (weights in [-127,127] are exact in f16; f16xf16 products are exact in the
// f32 accumulator), VALU only does int32->f16 conversion of the stream.
// x is staged once per block as f16 in LDS so the hot loop has no branches,
// no redundant global traffic, and no per-iteration zeroing.
// ---------------------------------------------------------------------------

typedef __attribute__((ext_vector_type(16))) _Float16 v16h;
typedef __attribute__((ext_vector_type(8)))  _Float16 v8h;
typedef __attribute__((ext_vector_type(8)))  float    v8f;
typedef __attribute__((ext_vector_type(4)))  int      v4i;
typedef __attribute__((ext_vector_type(4)))  float    v4f;

#define OUT_F 11008
#define IN_F  4096
#define BATCH 4

// Block: 128 threads = 4 waves. Each block owns a 16-channel tile of OUT_F.
// Wave w handles K slice [w*1024, (w+1)*1024); partials reduced via LDS in
// a fixed order (deterministic), then wave0 applies scale/bias and stores.
__global__ __launch_bounds__(128) void qlinear_wmma_f16(
    const float* __restrict__ x,       // [BATCH, IN_F]
    const int*   __restrict__ qw,      // [OUT_F, IN_F]
    const float* __restrict__ scales,  // [OUT_F]
    const float* __restrict__ bias,    // [OUT_F]
    float* __restrict__ out)           // [BATCH, OUT_F]
{
    __shared__ __align__(16) _Float16 xh[BATCH * IN_F]; // 32 KB: x in f16
    __shared__ __align__(16) _Float16 zpad[32];         // 64 B of zeros (M>=4 lanes)
    __shared__ float red[3][4][16];                     // split-K partials, waves 1..3

    const int tid  = threadIdx.x;
    const int lane = tid & 31;
    const int wave = tid >> 5;             // 0..3 : K split
    const int tile = blockIdx.x;           // 16-channel tile index
    const int n    = lane & 15;            // column within tile
    const int half = lane >> 4;            // lane half (K-half selector)

    // ---- stage x (fp32 -> f16) into LDS once per block ----------------------
    // 16384 elements, 8 per thread-step, 128 threads -> 16 steps, coalesced.
    for (int i = tid; i < (BATCH * IN_F) / 8; i += 128) {
        v4f p0 = *(const v4f*)(x + i * 8);
        v4f p1 = *(const v4f*)(x + i * 8 + 4);
        v8h h;
#pragma unroll
        for (int j = 0; j < 4; ++j) {
            h[j]     = (_Float16)p0[j];
            h[4 + j] = (_Float16)p1[j];
        }
        *(v8h*)(xh + i * 8) = h;
    }
    if (tid < 4) *(v8h*)(zpad + tid * 8) = (v8h){};
    __syncthreads();

    // ---- B operand (weights, 32x16): lane covers channel oc, 16 consecutive K
    //      (lanes 0-15: K 0..15 of the 32-chunk; lanes 16-31: K 16..31).
    const int oc = tile * 16 + n;
    const int* __restrict__ wrow = qw + (size_t)oc * IN_F;

    // ---- A operand (x, 16x32): lane covers batch row m = n (zero if m>=4);
    //      halves 0..15 = xh[m][k+16h .. +15].
    const int k0 = wave * (IN_F / 4);
    const int k1 = k0 + (IN_F / 4);
    const bool valid      = (n < BATCH);
    const _Float16* aptr  = valid ? (xh + n * IN_F + k0 + 16 * half) : zpad;
    const int       astep = valid ? 32 : 0;   // halfs advanced per 32-K iteration

    v8f acc = {};

#pragma unroll 2
    for (int k = k0; k < k1; k += 32) {
        // -------- weights: 16 consecutive int32 = 64 contiguous bytes/lane ----
        const int* wp = wrow + k + 16 * half;
        v4i wv0 = *(const v4i*)(wp + 0);
        v4i wv1 = *(const v4i*)(wp + 4);
        v4i wv2 = *(const v4i*)(wp + 8);
        v4i wv3 = *(const v4i*)(wp + 12);

        // stream hint for the next 32-K chunk of this row (into GL2)
        __builtin_prefetch((const void*)(wp + 32), 0, 0);

        v16h b;
#pragma unroll
        for (int j = 0; j < 4; ++j) {
            b[0  + j] = (_Float16)wv0[j];   // exact: |w| <= 127
            b[4  + j] = (_Float16)wv1[j];
            b[8  + j] = (_Float16)wv2[j];
            b[12 + j] = (_Float16)wv3[j];
        }

        // -------- activations from LDS: two 16B ds loads, no branches ---------
        v8h alo = *(const v8h*)(aptr);
        v8h ahi = *(const v8h*)(aptr + 8);
        aptr += astep;
        v16h a = __builtin_shufflevector(alo, ahi,
                                         0, 1, 2, 3, 4, 5, 6, 7,
                                         8, 9, 10, 11, 12, 13, 14, 15);

        // D = A(16x32 f16) x B(32x16 f16) + C(16x16 f32)
        acc = __builtin_amdgcn_wmma_f32_16x16x32_f16(
            /*neg_a=*/false, a, /*neg_b=*/false, b,
            /*c_mod=*/(short)0, acc, /*reuse_a=*/false, /*reuse_b=*/false);
    }

    // ---- deterministic split-K reduction over the 4 waves -------------------
    // C layout: acc[r], lanes 0-15  -> (M=r, N=lane); only M=0..3 are valid.
    if (wave != 0 && lane < 16) {
#pragma unroll
        for (int r = 0; r < BATCH; ++r)
            red[wave - 1][r][lane] = acc[r];
    }
    __syncthreads();

    if (wave == 0 && lane < 16) {
        const int   o  = tile * 16 + lane;
        const float s  = scales[o];
        const float bz = bias[o];
#pragma unroll
        for (int r = 0; r < BATCH; ++r) {
            float v = acc[r] + red[0][r][lane] + red[1][r][lane] + red[2][r][lane];
            out[(size_t)r * OUT_F + o] = s * v + bz;
        }
    }
}

extern "C" void kernel_launch(void* const* d_in, const int* in_sizes, int n_in,
                              void* d_out, int out_size, void* d_ws, size_t ws_size,
                              hipStream_t stream) {
    const float* x      = (const float*)d_in[0];
    const int*   qw     = (const int*)  d_in[1];
    const float* scales = (const float*)d_in[2];
    const float* bias   = (const float*)d_in[3];
    float*       out    = (float*)d_out;

    dim3 grid(OUT_F / 16);   // 688 tiles
    dim3 block(128);         // 4 waves: 4-way split-K per tile
    hipLaunchKernelGGL(qlinear_wmma_f16, grid, block, 0, stream,
                       x, qw, scales, bias, out);
}
